// InputFeedRNNDecoder_61692910240270
// MI455X (gfx1250) — matrix operand order repaired and checked by reading, not compiled, not run on hardware
//
#include <hip/hip_runtime.h>
#include <hip/hip_bf16.h>

// ---------------- CDNA5 WMMA types ----------------
typedef __attribute__((ext_vector_type(16))) __bf16 v16bf;
typedef __attribute__((ext_vector_type(8)))  float  v8f;

#define TT 100
#define BB 64
#define SS 602
#define HH 1024
#define EE 512
#define VV 10000

// WGP-scope prefetch (scope field 0 = default): pulls the line into all cache
// levels, unlike __builtin_prefetch which only reaches SE/SYS scope (GL2-only).
__device__ __forceinline__ void prefetch_wgp(const void* p)
{
    asm volatile("global_prefetch_b8 %0, off offset:1024" :: "v"(p));
}

// ---------------------------------------------------------------------------
// GEMM: C[64 x N] = X[64 x K] * W[N x K]^T  (+ bias0 + bias1), bf16 in, f32 out
// Block = 256 threads (8 waves). Block computes a 64 x 64 tile of C.
// Wave w: rows (w&3)*16..+15, cols blockIdx.x*64 + (w>>2)*32 .. +31
// K must be a multiple of 128 (holds for 2560 / 2048 / 1024).
// ---------------------------------------------------------------------------
__global__ __launch_bounds__(256) void gemm_bf16_wmma(
    const __bf16* __restrict__ X, const __bf16* __restrict__ W,
    const float* __restrict__ bias0, const float* __restrict__ bias1,
    float* __restrict__ C, int N, int K)
{
    const int tid      = threadIdx.x;
    const int wave     = tid >> 5;
    const int lane     = tid & 31;
    const int lane16   = lane & 15;
    const int laneHalf = lane >> 4;                 // 0 or 1

    const int mBase = (wave & 3) * 16;
    const int nBase = blockIdx.x * 64 + (wave >> 2) * 32;

    v8f acc0 = {};
    v8f acc1 = {};

    // A fragment (16-bit A 16x32): lanes 0-15 hold K {0..7} in v0..3, {16..23} in v4..7;
    // lanes 16-31 hold K {8..15} and {24..31}.
    const int aOff = laneHalf * 8;
    // B fragment (16-bit B 32x16): same K order as A, column lane16:
    // lanes 0-15 -> K {0..7},{16..23}; lanes 16-31 -> K {8..15},{24..31}.
    const int bOff = laneHalf * 8;

    const __bf16* __restrict__ Xrow = X + (size_t)(mBase + lane16) * K;
    const __bf16* __restrict__ Wn0  = W + (size_t)(nBase + lane16) * K;
    const __bf16* __restrict__ Wn1  = W + (size_t)(nBase + 16 + lane16) * K;

    for (int k0 = 0; k0 < K; k0 += 128) {
        // One WGP-scope prefetch per 256B of streamed W, 1KB ahead. Kept out
        // of the unrolled hot body so the asm cannot break the load pipeline.
        prefetch_wgp(Wn0 + k0);
        prefetch_wgp(Wn1 + k0);
        #pragma unroll
        for (int kk = 0; kk < 128; kk += 32) {
            const int k = k0 + kk;
            union { v16bf v; uint4 q[2]; } a, b0, b1;
            a.q[0]  = *(const uint4*)(Xrow + k + aOff);
            a.q[1]  = *(const uint4*)(Xrow + k + 16 + aOff);
            b0.q[0] = *(const uint4*)(Wn0 + k + bOff);
            b0.q[1] = *(const uint4*)(Wn0 + k + bOff + 16);
            b1.q[0] = *(const uint4*)(Wn1 + k + bOff);
            b1.q[1] = *(const uint4*)(Wn1 + k + bOff + 16);
            acc0 = __builtin_amdgcn_wmma_f32_16x16x32_bf16(false, a.v, false, b0.v,
                                                           (short)0, acc0, false, false);
            acc1 = __builtin_amdgcn_wmma_f32_16x16x32_bf16(false, a.v, false, b1.v,
                                                           (short)0, acc1, false, false);
        }
    }

    // C/D layout: VGPR r -> M = mBase + r + 8*laneHalf, N = nBase + lane16 (+16 for acc1)
    #pragma unroll
    for (int r = 0; r < 8; ++r) {
        const int row  = mBase + laneHalf * 8 + r;
        const int col0 = nBase + lane16;
        const int col1 = col0 + 16;
        float b0s = 0.f, b1s = 0.f;
        if (bias0) { b0s += bias0[col0]; b1s += bias0[col1]; }
        if (bias1) { b0s += bias1[col0]; b1s += bias1[col1]; }
        C[(size_t)row * N + col0] = acc0[r] + b0s;
        C[(size_t)row * N + col1] = acc1[r] + b1s;
    }
}

// ---------------------------------------------------------------------------
// Pack [A | B] (fp32, each N x Ka / N x Kb, row-major) into bf16 N x (Ka+Kb)
// ---------------------------------------------------------------------------
__global__ void pack2_bf16(__bf16* __restrict__ dst,
                           const float* __restrict__ A, int Ka,
                           const float* __restrict__ Bsrc, int Kb, int total)
{
    int i = blockIdx.x * 256 + threadIdx.x;
    if (i >= total) return;
    const int Kt = Ka + Kb;
    const int n = i / Kt;
    const int k = i % Kt;
    float v = (k < Ka) ? A[(size_t)n * Ka + k] : Bsrc[(size_t)n * Kb + (k - Ka)];
    dst[i] = (__bf16)v;
}

// fp32 -> bf16 flat conversion (used for the 158MB memory bank -> 79MB,
// keeping the whole working set resident in the 192MB L2).
__global__ void f2bf_flat(__bf16* __restrict__ dst, const float* __restrict__ src, int total)
{
    int i = blockIdx.x * 256 + threadIdx.x;
    if (i >= total) return;
    dst[i] = (__bf16)src[i];
}

// Copy LSTM state in from inputs, zero the input-feed vector.
__global__ void init_state(const float* __restrict__ h0, const float* __restrict__ c0,
                           float* h0s, float* h1s, float* c0s, float* c1s, float* feed)
{
    int i = blockIdx.x * 256 + threadIdx.x;
    if (i >= BB * HH) return;
    h0s[i] = h0[i];
    h1s[i] = h0[BB * HH + i];
    c0s[i] = c0[i];
    c1s[i] = c0[BB * HH + i];
    feed[i] = 0.f;
}

// xh0[b, :] = bf16([ emb(tgt[b]) | feed | h_l0 ])   (B x 2560)
__global__ void build_xh0(__bf16* __restrict__ xh0,
                          const float* __restrict__ emb, const int* __restrict__ tgt_t,
                          const float* __restrict__ feed, const float* __restrict__ h0s)
{
    int i = blockIdx.x * 256 + threadIdx.x;
    if (i >= BB * (EE + 2 * HH)) return;
    const int b = i / (EE + 2 * HH);
    const int k = i % (EE + 2 * HH);
    float v;
    if (k < EE)            v = emb[(size_t)tgt_t[b] * EE + k];
    else if (k < EE + HH)  v = feed[b * HH + (k - EE)];
    else                   v = h0s[b * HH + (k - EE - HH)];
    xh0[i] = (__bf16)v;
}

// Strided fp32 -> bf16 row copy (64 rows x HH cols) into a wider bf16 matrix.
__global__ void f2bf_rows(__bf16* __restrict__ dst, int dstStride, int dstOff,
                          const float* __restrict__ src)
{
    int i = blockIdx.x * 256 + threadIdx.x;
    if (i >= BB * HH) return;
    const int b = i / HH, k = i % HH;
    dst[b * dstStride + dstOff + k] = (__bf16)src[i];
}

// LSTM gate nonlinearity; updates c/h in place and mirrors new h as bf16 into
// up to two strided destinations (next-layer / attention GEMM operands).
__global__ void lstm_cell(const float* __restrict__ g, float* __restrict__ c,
                          float* __restrict__ h,
                          __bf16* bf0, int s0, int o0,
                          __bf16* bf1, int s1, int o1)
{
    int i = blockIdx.x * 256 + threadIdx.x;
    if (i >= BB * HH) return;
    const int b = i >> 10, k = i & (HH - 1);
    const float* gr = g + (size_t)b * 4 * HH;
    const float ig = gr[k], fg = gr[HH + k], gg = gr[2 * HH + k], og = gr[3 * HH + k];
    const float si = 1.f / (1.f + __expf(-ig));
    const float sf = 1.f / (1.f + __expf(-fg));
    const float so = 1.f / (1.f + __expf(-og));
    const float c2 = sf * c[i] + si * tanhf(gg);
    const float h2 = so * tanhf(c2);
    c[i] = c2;
    h[i] = h2;
    if (bf0) bf0[b * s0 + o0 + k] = (__bf16)h2;
    if (bf1) bf1[b * s1 + o1 + k] = (__bf16)h2;
}

// ---------------------------------------------------------------------------
// Attention score / context kernels — bf16-bank versions (L2-resident path)
// and fp32-bank fallbacks (used when ws is too small for the bf16 bank).
// ---------------------------------------------------------------------------
__global__ __launch_bounds__(256) void attn_scores_bf(const float* __restrict__ q,
                                                      const __bf16* __restrict__ mbb,
                                                      float* __restrict__ align)
{
    const int gw   = (blockIdx.x * 256 + threadIdx.x) >> 5;
    const int lane = threadIdx.x & 31;
    if (gw >= BB * SS) return;
    const int b = gw / SS, s = gw % SS;
    const float* __restrict__ qa    = q + (size_t)b * HH;
    const __bf16* __restrict__ mrow = mbb + ((size_t)s * BB + b) * HH;
    float sum = 0.f;
    for (int k = lane; k < HH; k += 32) sum += qa[k] * (float)mrow[k];
    #pragma unroll
    for (int off = 16; off > 0; off >>= 1) sum += __shfl_xor(sum, off, 32);
    if (lane == 0) align[b * SS + s] = sum;
}

__global__ __launch_bounds__(256) void attn_scores_f32(const float* __restrict__ q,
                                                       const float* __restrict__ mb,
                                                       float* __restrict__ align)
{
    const int gw   = (blockIdx.x * 256 + threadIdx.x) >> 5;
    const int lane = threadIdx.x & 31;
    if (gw >= BB * SS) return;
    const int b = gw / SS, s = gw % SS;
    const float* __restrict__ qa   = q + (size_t)b * HH;
    const float* __restrict__ mrow = mb + ((size_t)s * BB + b) * HH;
    float sum = 0.f;
    for (int k = lane; k < HH; k += 32) sum += qa[k] * mrow[k];
    #pragma unroll
    for (int off = 16; off > 0; off >>= 1) sum += __shfl_xor(sum, off, 32);
    if (lane == 0) align[b * SS + s] = sum;
}

// Row softmax over S; write normalized weights to ws buffer and to attns output.
__global__ __launch_bounds__(256) void softmax_rows(const float* __restrict__ align,
                                                    float* __restrict__ a_ws,
                                                    float* __restrict__ attns_t)
{
    __shared__ float red[256];
    const int b = blockIdx.x, t = threadIdx.x;
    const float* row = align + b * SS;
    float m = -1e30f;
    for (int s = t; s < SS; s += 256) m = fmaxf(m, row[s]);
    red[t] = m; __syncthreads();
    for (int o = 128; o > 0; o >>= 1) { if (t < o) red[t] = fmaxf(red[t], red[t + o]); __syncthreads(); }
    m = red[0]; __syncthreads();
    float sum = 0.f;
    for (int s = t; s < SS; s += 256) sum += __expf(row[s] - m);
    red[t] = sum; __syncthreads();
    for (int o = 128; o > 0; o >>= 1) { if (t < o) red[t] += red[t + o]; __syncthreads(); }
    const float inv = 1.f / red[0];
    for (int s = t; s < SS; s += 256) {
        const float v = __expf(row[s] - m) * inv;
        a_ws[b * SS + s]    = v;
        attns_t[b * SS + s] = v;
    }
}

// ctx[b,h] = sum_s a[b,s] * M[s,b,h]; write bf16 into ctxnh[:, 0:H] (stride 2H)
__global__ __launch_bounds__(256) void attn_ctx_bf(const float* __restrict__ a,
                                                   const __bf16* __restrict__ mbb,
                                                   __bf16* __restrict__ ctxnh)
{
    const int b = blockIdx.x, t = threadIdx.x;
    float acc[4] = {0.f, 0.f, 0.f, 0.f};
    for (int s = 0; s < SS; ++s) {
        const float w = a[b * SS + s];
        const __bf16* __restrict__ mrow = mbb + ((size_t)s * BB + b) * HH;
        #pragma unroll
        for (int j = 0; j < 4; ++j) acc[j] += w * (float)mrow[t + j * 256];
    }
    #pragma unroll
    for (int j = 0; j < 4; ++j)
        ctxnh[b * 2 * HH + t + j * 256] = (__bf16)acc[j];
}

__global__ __launch_bounds__(256) void attn_ctx_f32(const float* __restrict__ a,
                                                    const float* __restrict__ mb,
                                                    __bf16* __restrict__ ctxnh)
{
    const int b = blockIdx.x, t = threadIdx.x;
    float acc[4] = {0.f, 0.f, 0.f, 0.f};
    for (int s = 0; s < SS; ++s) {
        const float w = a[b * SS + s];
        const float* __restrict__ mrow = mb + ((size_t)s * BB + b) * HH;
        #pragma unroll
        for (int j = 0; j < 4; ++j) acc[j] += w * mrow[t + j * 256];
    }
    #pragma unroll
    for (int j = 0; j < 4; ++j)
        ctxnh[b * 2 * HH + t + j * 256] = (__bf16)acc[j];
}

// out = tanh(gemm result); write fp32 to outs[t] and to next-step feed buffer
__global__ void tanh_out(const float* __restrict__ gout, float* __restrict__ outs_t,
                         float* __restrict__ feed)
{
    int i = blockIdx.x * 256 + threadIdx.x;
    if (i >= BB * HH) return;
    const float v = tanhf(gout[i]);
    outs_t[i] = v;
    feed[i]   = v;
}

// ---------------------------------------------------------------------------
extern "C" void kernel_launch(void* const* d_in, const int* in_sizes, int n_in,
                              void* d_out, int out_size, void* d_ws, size_t ws_size,
                              hipStream_t stream)
{
    (void)in_sizes; (void)n_in; (void)out_size;
    const int*   tgt    = (const int*)  d_in[0];   // (T,B,1)
    const float* mb     = (const float*)d_in[1];   // (S,B,H)
    const float* h0in   = (const float*)d_in[2];   // (2,B,H)
    const float* c0in   = (const float*)d_in[3];   // (2,B,H)
    const float* embed  = (const float*)d_in[4];   // (V,E)
    const float* Wi0    = (const float*)d_in[5];   // (4H, E+H)
    const float* Wh0    = (const float*)d_in[6];   // (4H, H)
    const float* bi0    = (const float*)d_in[7];
    const float* bh0    = (const float*)d_in[8];
    const float* Wi1    = (const float*)d_in[9];   // (4H, H)
    const float* Wh1    = (const float*)d_in[10];  // (4H, H)
    const float* bi1    = (const float*)d_in[11];
    const float* bh1    = (const float*)d_in[12];
    const float* Wattn  = (const float*)d_in[13];  // (H, H)
    const float* Wout   = (const float*)d_in[14];  // (H, 2H)

    float* outs  = (float*)d_out;                        // (T,B,H)
    float* attns = (float*)d_out + (size_t)TT * BB * HH; // (T,B,S)

    // ---- workspace carve-up (all 256B aligned) ----
    char* p = (char*)d_ws;
    auto carve = [&](size_t bytes) { void* r = (void*)p; p += (bytes + 255) & ~(size_t)255; return r; };
    const int K0 = EE + 2 * HH;   // 2560
    const int K1 = 2 * HH;        // 2048
    __bf16* Wcat0  = (__bf16*)carve((size_t)4 * HH * K0 * 2);
    __bf16* Wcat1  = (__bf16*)carve((size_t)4 * HH * K1 * 2);
    __bf16* WattnB = (__bf16*)carve((size_t)HH * HH * 2);
    __bf16* WoutB  = (__bf16*)carve((size_t)HH * K1 * 2);
    __bf16* xh0    = (__bf16*)carve((size_t)BB * K0 * 2);
    __bf16* xh1    = (__bf16*)carve((size_t)BB * K1 * 2);
    __bf16* qin    = (__bf16*)carve((size_t)BB * HH * 2);
    __bf16* ctxnh  = (__bf16*)carve((size_t)BB * K1 * 2);
    float*  g0     = (float*) carve((size_t)BB * 4 * HH * 4);
    float*  g1     = (float*) carve((size_t)BB * 4 * HH * 4);
    float*  qf     = (float*) carve((size_t)BB * HH * 4);
    float*  alignb = (float*) carve((size_t)BB * SS * 4);
    float*  aws    = (float*) carve((size_t)BB * SS * 4);
    float*  og     = (float*) carve((size_t)BB * HH * 4);
    float*  h0s    = (float*) carve((size_t)BB * HH * 4);
    float*  h1s    = (float*) carve((size_t)BB * HH * 4);
    float*  c0s    = (float*) carve((size_t)BB * HH * 4);
    float*  c1s    = (float*) carve((size_t)BB * HH * 4);
    float*  feed   = (float*) carve((size_t)BB * HH * 4);
    // bf16 memory bank last; only used if ws is big enough for it.
    __bf16* mbb    = (__bf16*)carve((size_t)SS * BB * HH * 2);
    const bool useBf16Bank = ((size_t)(p - (char*)d_ws) <= ws_size);

    // ---- one-time (per launch) weight packing to bf16 ----
    {
        int tot0 = 4 * HH * K0;
        pack2_bf16<<<(tot0 + 255) / 256, 256, 0, stream>>>(Wcat0, Wi0, EE + HH, Wh0, HH, tot0);
        int tot1 = 4 * HH * K1;
        pack2_bf16<<<(tot1 + 255) / 256, 256, 0, stream>>>(Wcat1, Wi1, HH, Wh1, HH, tot1);
        int tota = HH * HH;
        pack2_bf16<<<(tota + 255) / 256, 256, 0, stream>>>(WattnB, Wattn, HH, nullptr, 0, tota);
        int toto = HH * K1;
        pack2_bf16<<<(toto + 255) / 256, 256, 0, stream>>>(WoutB, Wout, K1, nullptr, 0, toto);
        init_state<<<(BB * HH + 255) / 256, 256, 0, stream>>>(h0in, c0in, h0s, h1s, c0s, c1s, feed);
        if (useBf16Bank) {
            int totm = SS * BB * HH;
            f2bf_flat<<<(totm + 255) / 256, 256, 0, stream>>>(mbb, mb, totm);
        }
    }

    const dim3 blk(256);
    const int ew = (BB * HH + 255) / 256;     // elementwise grid over B*H

    for (int t = 0; t < TT; ++t) {
        const int* tgt_t = tgt + (size_t)t * BB;

        // layer 0: xh0 = [emb | feed | h_l0], g0 = xh0 * [Wi0|Wh0]^T + bi0 + bh0
        build_xh0<<<(BB * K0 + 255) / 256, blk, 0, stream>>>(xh0, embed, tgt_t, feed, h0s);
        gemm_bf16_wmma<<<4 * HH / 64, blk, 0, stream>>>(xh0, Wcat0, bi0, bh0, g0, 4 * HH, K0);
        // cell 0 -> updates (h0s, c0s), drops bf16(nh0) into xh1[:, 0:H]
        lstm_cell<<<ew, blk, 0, stream>>>(g0, c0s, h0s, xh1, K1, 0, nullptr, 0, 0);
        // xh1[:, H:2H] = bf16(old h_l1)
        f2bf_rows<<<ew, blk, 0, stream>>>(xh1, K1, HH, h1s);

        // layer 1
        gemm_bf16_wmma<<<4 * HH / 64, blk, 0, stream>>>(xh1, Wcat1, bi1, bh1, g1, 4 * HH, K1);
        // cell 1 -> updates (h1s, c1s), drops bf16(nh1) into qin and ctxnh[:, H:2H]
        lstm_cell<<<ew, blk, 0, stream>>>(g1, c1s, h1s, qin, HH, 0, ctxnh, K1, HH);

        // attention
        gemm_bf16_wmma<<<HH / 64, blk, 0, stream>>>(qin, WattnB, nullptr, nullptr, qf, HH, HH);
        if (useBf16Bank)
            attn_scores_bf<<<(BB * SS) / 8, blk, 0, stream>>>(qf, mbb, alignb);
        else
            attn_scores_f32<<<(BB * SS) / 8, blk, 0, stream>>>(qf, mb, alignb);
        softmax_rows<<<BB, blk, 0, stream>>>(alignb, aws, attns + (size_t)t * BB * SS);
        if (useBf16Bank)
            attn_ctx_bf<<<BB, blk, 0, stream>>>(aws, mbb, ctxnh);
        else
            attn_ctx_f32<<<BB, blk, 0, stream>>>(aws, mb, ctxnh);

        // output projection + tanh; refresh input feed
        gemm_bf16_wmma<<<HH / 64, blk, 0, stream>>>(ctxnh, WoutB, nullptr, nullptr, og, HH, K1);
        tanh_out<<<ew, blk, 0, stream>>>(og, outs + (size_t)t * BB * HH, feed);
    }
}
